// DeformConv_15427522527490
// MI455X (gfx1250) — matrix operand order repaired and checked by reading, not compiled, not run on hardware
//
#include <hip/hip_runtime.h>

#define Bsz 4
#define Csz 64
#define Osz 64
#define Hsz 128
#define Wsz 128
#define HW  (Hsz * Wsz)          // 16384
#define Ksz 576                  // 9 * 64
#define AP  592                  // padded A-tile pitch in halves (16B aligned)
#define Mtot (Bsz * HW)          // 65536

typedef __attribute__((ext_vector_type(16))) __bf16 v16bf;
typedef __attribute__((ext_vector_type(8)))  float  v8f;

__device__ __forceinline__ unsigned short f2bf(float f) {
    union { float f; unsigned u; } v; v.f = f;
    unsigned r = v.u + 0x7FFFu + ((v.u >> 16) & 1u);   // round to nearest even
    return (unsigned short)(r >> 16);
}

// ---------------- kernel 0: zero BN accumulators ----------------
__global__ void k_zero(float* stats) {
    if (threadIdx.x < 128) stats[threadIdx.x] = 0.0f;
}

// ---------------- kernel 0b: NCHW -> NHWC transpose of x ----------------
// i = (b*HW + hw)*64 + c ; write coalesced (c fastest), reads L2-absorbed.
__global__ void __launch_bounds__(256) k_nhwc(
    const float* __restrict__ x, float* __restrict__ xn) {
    int i = blockIdx.x * 256 + threadIdx.x;          // 0 .. B*HW*64-1
    int c  = i & 63;
    int bhw = i >> 6;
    int b  = bhw >> 14;                               // / HW
    int hw = bhw & (HW - 1);
    xn[i] = x[((size_t)b * Csz + c) * HW + hw];
}

// ---------------- kernel 1: offset-predicting 3x3 conv ----------------
// grid: (HW/256, 27, B), block 256. Weight index is block-uniform -> scalar loads.
__global__ void __launch_bounds__(256) k_offconv(
    const float* __restrict__ x, const float* __restrict__ ow,
    const float* __restrict__ ob, float* __restrict__ om) {
    const int hw = blockIdx.x * 256 + threadIdx.x;
    const int j  = blockIdx.y;
    const int b  = blockIdx.z;
    const int h = hw >> 7, w = hw & 127;
    float acc = ob[j];
    const float* xb = x + (size_t)b * Csz * HW;
    const float* wj = ow + (size_t)j * Csz * 9;
    for (int c = 0; c < Csz; ++c) {
        const float* xc = xb + c * HW;
        const float* wc = wj + c * 9;
#pragma unroll
        for (int ky = 0; ky < 3; ++ky) {
            int yy = h + ky - 1;
            if (yy < 0 || yy >= Hsz) continue;
#pragma unroll
            for (int kx = 0; kx < 3; ++kx) {
                int xx = w + kx - 1;
                if (xx < 0 || xx >= Wsz) continue;
                acc = fmaf(xc[yy * Wsz + xx], wc[ky * 3 + kx], acc);
            }
        }
    }
    om[((size_t)b * 27 + j) * HW + hw] = acc;
}

// ---------------- kernel 2: repack dcn_w -> bf16 B^T [O][K], K = kk*64 + c ----
__global__ void k_wprep(const float* __restrict__ dw, unsigned short* __restrict__ Bw) {
    int i = blockIdx.x * blockDim.x + threadIdx.x;
    if (i >= Osz * Ksz) return;
    int o = i / Ksz, k = i - o * Ksz;
    int kk = k >> 6, c = k & 63;
    Bw[i] = f2bf(dw[(o * Csz + c) * 9 + kk]);
}

// ---------------- kernel 3: fused deformable-sample + WMMA GEMM ----------------
// xs: activation tensor; element (b,c,idx) at xs[b*Csz*HW + c*cstride + idx*istride]
//   NHWC: cstride=1,  istride=64   (coalesced gathers: lanes = consecutive c)
//   NCHW: cstride=HW, istride=1    (fallback when workspace too small)
struct Meta { int i00, i01, i10, i11; float w00, w01, w10, w11; };

__global__ void __launch_bounds__(256) k_dcn(
    const float* __restrict__ xs, int cstride, int istride,
    const float* __restrict__ om,
    const unsigned short* __restrict__ Bw, const float* __restrict__ dcnb,
    float* __restrict__ outp, float* __restrict__ stats) {
    __shared__ Meta meta[32 * 9];
    __shared__ __align__(16) unsigned short Atile[32 * AP];
    __shared__ float rsum[Osz], rsumsq[Osz];

    const int tid = threadIdx.x;
    const int m0  = blockIdx.x * 32;         // 32 pixels per block; b uniform in tile
    const int b   = m0 / HW;
    const int hw0 = m0 - b * HW;

    if (tid < Osz) { rsum[tid] = 0.0f; rsumsq[tid] = 0.0f; }

    // ---- phase 1: bilinear metadata for 32 pixels x 9 taps ----
    for (int t = tid; t < 288; t += 256) {
        int p = t / 9, kk = t - p * 9;
        int hw = hw0 + p;
        int h = hw >> 7, w = hw & 127;
        int ky = kk / 3, kx = kk - ky * 3;
        const float* omb = om + (size_t)b * 27 * HW + hw;
        float offx = omb[(size_t)kk * HW];
        float offy = omb[(size_t)(9 + kk) * HW];
        float mk   = omb[(size_t)(18 + kk) * HW];
        mk = 1.0f / (1.0f + __expf(-mk));             // sigmoid modulation
        float py = (float)(h - 1 + ky) + offy;
        float px = (float)(w - 1 + kx) + offx;
        float fy = floorf(py), fx = floorf(px);
        float wy = py - fy, wx = px - fx;
        int y0 = (int)fy, x0 = (int)fx, y1 = y0 + 1, x1 = x0 + 1;
        bool vy0 = (y0 >= 0) & (y0 < Hsz), vy1 = (y1 >= 0) & (y1 < Hsz);
        bool vx0 = (x0 >= 0) & (x0 < Wsz), vx1 = (x1 >= 0) & (x1 < Wsz);
        int cy0 = min(max(y0, 0), Hsz - 1), cy1 = min(max(y1, 0), Hsz - 1);
        int cx0 = min(max(x0, 0), Wsz - 1), cx1 = min(max(x1, 0), Wsz - 1);
        Meta mt;
        mt.i00 = (cy0 * Wsz + cx0) * istride;  mt.i01 = (cy0 * Wsz + cx1) * istride;
        mt.i10 = (cy1 * Wsz + cx0) * istride;  mt.i11 = (cy1 * Wsz + cx1) * istride;
        mt.w00 = (vy0 && vx0) ? (1.0f - wy) * (1.0f - wx) * mk : 0.0f;
        mt.w01 = (vy0 && vx1) ? (1.0f - wy) * wx * mk : 0.0f;
        mt.w10 = (vy1 && vx0) ? wy * (1.0f - wx) * mk : 0.0f;
        mt.w11 = (vy1 && vx1) ? wy * wx * mk : 0.0f;
        meta[t] = mt;
    }
    __syncthreads();

    // ---- phase 2: build bf16 A tile [32 pixels][K=576] in LDS ----
    // NHWC path: lanes (c) read consecutive floats -> 2 coalesced lines per corner.
    {
        const int c = tid & 63;
        const float* xc = xs + (size_t)b * Csz * HW + (size_t)c * cstride;
        for (int e = tid >> 6; e < 288; e += 4) {
            Meta mt = meta[e];
            float v = mt.w00 * xc[mt.i00] + mt.w01 * xc[mt.i01]
                    + mt.w10 * xc[mt.i10] + mt.w11 * xc[mt.i11];
            int p = e / 9, kk = e - p * 9;
            Atile[p * AP + (kk << 6) + c] = f2bf(v);
        }
    }
    __syncthreads();

    // ---- phase 3: WMMA main loop (each wave: 16 pixels x 16 channels) ----
    const int wave = tid >> 5, lane = tid & 31;
    const int ti = wave >> 2;                 // pixel sub-tile 0..1
    const int tj = wave & 3;                  // channel tile 0..3
    const int mrow = lane & 15;
    const int k0   = lane >> 4;
    const unsigned short* arow = Atile + (ti * 16 + mrow) * AP;
    const unsigned short* brow = Bw + (size_t)(tj * 16 + mrow) * Ksz;
    v8f acc = {0.f, 0.f, 0.f, 0.f, 0.f, 0.f, 0.f, 0.f};
#pragma unroll
    for (int ks = 0; ks < 18; ++ks) {
        union { v16bf v; uint4 u[2]; } A, Bf;
        // A 16x32 bf16 layout: lanes 0-15 K{0..7,16..23}, lanes 16-31 K{8..15,24..31}
        const uint4* ap = (const uint4*)(arow + ks * 32);
        A.u[0] = ap[k0];
        A.u[1] = ap[2 + k0];
        // B 32x16 bf16 layout: column = lane&15; lanes 0-15 K{0..7,16..23}, lanes 16-31 K{8..15,24..31}
        const uint4* bp = (const uint4*)(brow + ks * 32);
        Bf.u[0] = bp[k0];
        Bf.u[1] = bp[2 + k0];
        acc = __builtin_amdgcn_wmma_f32_16x16x32_bf16(
            false, A.v, false, Bf.v, (short)0, acc, false, false);
    }

    // ---- epilogue: +bias, store pre-BN, per-channel stats ----
    const int o = tj * 16 + mrow;
    const float bias = dcnb[o];
    const int hwL = hw0 + ti * 16 + (k0 << 3);   // VGPR r: M = r (+8 for upper lanes)
    float s = 0.0f, ss = 0.0f;
    float* op = outp + ((size_t)b * Osz + o) * HW;
#pragma unroll
    for (int r = 0; r < 8; ++r) {
        float v = acc[r] + bias;
        op[hwL + r] = v;
        s += v; ss += v * v;
    }
    __syncthreads();
    atomicAdd(&rsum[o], s);
    atomicAdd(&rsumsq[o], ss);
    __syncthreads();
    if (tid < Osz) {
        atomicAdd(&stats[tid], rsum[tid]);
        atomicAdd(&stats[Osz + tid], rsumsq[tid]);
    }
}

// ---------------- kernel 4: BN stats -> scale/shift ----------------
__global__ void k_bnfin(const float* __restrict__ stats, const float* __restrict__ gamma,
                        const float* __restrict__ beta, float* __restrict__ sc) {
    int o = threadIdx.x;
    if (o >= Osz) return;
    const float invN = 1.0f / (float)Mtot;
    float mean = stats[o] * invN;
    float var  = stats[Osz + o] * invN - mean * mean;
    float scale = gamma[o] * rsqrtf(var + 1e-5f);
    sc[o] = scale;
    sc[Osz + o] = beta[o] - mean * scale;
}

// ---------------- kernel 5: apply BN + ReLU ----------------
__global__ void __launch_bounds__(256) k_bnapply(
    const float* __restrict__ outp, const float* __restrict__ sc,
    float* __restrict__ out) {
    int i = blockIdx.x * 256 + threadIdx.x;
    int o = (i >> 14) & 63;                  // (i / HW) % O
    float v = outp[i] * sc[o] + sc[Osz + o];
    out[i] = v > 0.0f ? v : 0.0f;
}

extern "C" void kernel_launch(void* const* d_in, const int* in_sizes, int n_in,
                              void* d_out, int out_size, void* d_ws, size_t ws_size,
                              hipStream_t stream) {
    (void)in_sizes; (void)n_in; (void)out_size;
    const float* x     = (const float*)d_in[0];
    const float* ow    = (const float*)d_in[1];
    const float* ob    = (const float*)d_in[2];
    const float* dw    = (const float*)d_in[3];
    const float* db    = (const float*)d_in[4];
    const float* gamma = (const float*)d_in[5];
    const float* beta  = (const float*)d_in[6];

    char* ws = (char*)d_ws;
    size_t off = 0;
    float* om = (float*)(ws + off);           off += (size_t)Bsz * 27 * HW * sizeof(float);
    float* outp = (float*)(ws + off);         off += (size_t)Bsz * Osz * HW * sizeof(float);
    unsigned short* Bw = (unsigned short*)(ws + off);
                                              off += (size_t)Osz * Ksz * sizeof(unsigned short);
    off = (off + 255) & ~(size_t)255;
    float* stats = (float*)(ws + off);        off += 128 * sizeof(float);
    float* sc = (float*)(ws + off);           off += 128 * sizeof(float);
    off = (off + 255) & ~(size_t)255;
    float* xn = (float*)(ws + off);           off += (size_t)Bsz * Csz * HW * sizeof(float);
    const bool use_nhwc = (ws_size >= off);   // fixed per process -> deterministic path

    k_zero<<<1, 128, 0, stream>>>(stats);
    k_offconv<<<dim3(HW / 256, 27, Bsz), 256, 0, stream>>>(x, ow, ob, om);
    k_wprep<<<(Osz * Ksz + 255) / 256, 256, 0, stream>>>(dw, Bw);
    if (use_nhwc) {
        k_nhwc<<<(Bsz * Csz * HW) / 256, 256, 0, stream>>>(x, xn);
        k_dcn<<<Mtot / 32, 256, 0, stream>>>(xn, /*cstride=*/1, /*istride=*/Csz,
                                             om, Bw, db, outp, stats);
    } else {
        k_dcn<<<Mtot / 32, 256, 0, stream>>>(x, /*cstride=*/HW, /*istride=*/1,
                                             om, Bw, db, outp, stats);
    }
    k_bnfin<<<1, 64, 0, stream>>>(stats, gamma, beta, sc);
    k_bnapply<<<(Bsz * Osz * HW) / 256, 256, 0, stream>>>(outp, sc, (float*)d_out);
}
